// OpenaiExperts_29875792511352
// MI455X (gfx1250) — matrix operand order repaired and checked by reading, not compiled, not run on hardware
//
#include <hip/hip_runtime.h>
#include <hip/hip_bf16.h>

typedef __attribute__((ext_vector_type(16))) __bf16       v16bf;
typedef __attribute__((ext_vector_type(8)))  float        v8f;
typedef __attribute__((ext_vector_type(4)))  float        f32x4;
typedef __attribute__((ext_vector_type(4)))  unsigned int u32x4;
typedef __attribute__((ext_vector_type(2)))  unsigned int u32x2;
typedef __attribute__((ext_vector_type(8)))  unsigned int u32x8;

constexpr int EXPERTS = 8;
constexpr int TOK     = 1024;
constexpr int HID     = 2880;
constexpr int INTER   = 2880;
constexpr int TWOI    = 5760;
constexpr int BK      = 32;        // K per WMMA step (16x16x32 bf16)
constexpr int NK      = HID / BK;  // 90
constexpr float ALPHA = 1.702f;

union FragU { v16bf v; u32x4 u[2]; };

// Hardware packed conversion: D[15:0]=bf16(lo), D[31:16]=bf16(hi)  (1 VALU op)
static __device__ __forceinline__ unsigned int cvt_pk_bf16(float lo, float hi) {
    unsigned int r;
    asm("v_cvt_pk_bf16_f32 %0, %1, %2" : "=v"(r) : "v"(lo), "v"(hi));
    return r;
}

static __device__ __forceinline__ unsigned short f2bf(float f) {
    return (unsigned short)(cvt_pk_bf16(f, 0.0f) & 0xFFFFu);
}

// A tile in LDS: row-major [M][32] bf16 (64B rows). ISA 16-bit A 16x32 layout:
// lanes 0-15: K0-7 (unit0) + K16-23 (unit2); lanes 16-31: K8-15 (unit1) + K24-31 (unit3).
static __device__ __forceinline__ v16bf load_afrag(const unsigned short* ldsA, int mbase, int lid) {
    int row = mbase + (lid & 15);
    int hi  = lid >> 4;
    const u32x4* p = (const u32x4*)(ldsA + row * BK);
    FragU f;
    f.u[0] = p[hi];
    f.u[1] = p[hi + 2];
    return f.v;
}

// B tile in LDS: transposed [N][32] bf16. ISA 16-bit B 32x16 layout:
// lanes 0-15: col=lane, K0-7 (unit0) + K16-23 (unit2); lanes 16-31: col=lane-16, K8-15 (unit1) + K24-31 (unit3).
static __device__ __forceinline__ v16bf load_bfrag(const unsigned short* ldsB, int nbase, int lid) {
    int row = nbase + (lid & 15);
    int hi  = lid >> 4;
    const u32x4* p = (const u32x4*)(ldsB + row * BK);
    FragU f;
    f.u[0] = p[hi];
    f.u[1] = p[hi + 2];
    return f.v;
}

#define WMMA_BF16(A, B, C) \
    __builtin_amdgcn_wmma_f32_16x16x32_bf16(false, (A), false, (B), (short)0, (C), false, false)

// ---------------------------------------------------------------------------
// TDM: DMA a 2D bf16 tile (rows x cols, row-major, stride elems) into LDS.
// Descriptor layout per CDNA5 ISA ch.8 (D# groups 0/1; groups 2/3 zero = 2D).
// Issued per-wave (EXEC ignored); caller gates to one wave per block and
// must s_wait_tensorcnt before the workgroup barrier.
// ---------------------------------------------------------------------------
static __device__ __forceinline__ void tdm_load_tile_bf16(const unsigned short* gsrc,
                                                          unsigned lds_byte_off,
                                                          unsigned stride_elems,
                                                          unsigned rows, unsigned cols)
{
    unsigned long long ga = (unsigned long long)(uintptr_t)gsrc;
    u32x4 g0;
    g0.x = 1u;                                                  // count=1, user mode
    g0.y = lds_byte_off;                                        // lds_addr
    g0.z = (unsigned)ga;                                        // global_addr[31:0]
    g0.w = (unsigned)((ga >> 32) & 0x1FFFFFFu) | (2u << 30);    // global_addr[56:32] | type=2
    u32x8 g1;
    g1[0] = 1u << 16;                                           // data_size=1 (2B), wg_mask=0
    g1[1] = (stride_elems & 0xFFFFu) << 16;                     // tensor_dim0[15:0]
    g1[2] = ((stride_elems >> 16) & 0xFFFFu) | ((rows & 0xFFFFu) << 16); // dim0 hi | tensor_dim1 lo
    g1[3] = ((rows >> 16) & 0xFFFFu) | ((cols & 0xFFFFu) << 16);         // dim1 hi | tile_dim0
    g1[4] = rows & 0xFFFFu;                                     // tile_dim1 = rows, tile_dim2 = 0
    g1[5] = stride_elems;                                       // tensor_dim0_stride[31:0]
    g1[6] = 0u;                                                 // stride hi | dim1_stride lo
    g1[7] = 0u;
    u32x4 gz = { 0u, 0u, 0u, 0u };
    asm volatile("tensor_load_to_lds %0, %1, %2, %3"
                 :: "s"(g0), "s"(g1), "s"(gz), "s"(gz)
                 : "memory");
}

// ---------------------------------------------------------------------------
// Kernel 1: gate_up GEMM + bias + swiglu, fused. Writes bf16 intermediate.
// Block: 256 thr (8 waves). Tile: 128 rows x 64 cols (per half).
// Double-buffered LDS; global loads for tile kt+1 issued before WMMAs of kt.
// ---------------------------------------------------------------------------
__global__ __launch_bounds__(256)
void gate_up_swiglu_kernel(const float* __restrict__ hs,
                           const float* __restrict__ gup,
                           const float* __restrict__ gub,
                           unsigned short* __restrict__ swbuf)
{
    __shared__ __align__(16) unsigned short ldsA[2][128 * BK];
    __shared__ __align__(16) unsigned short ldsB[2][128 * BK]; // rows 0-63 glu, 64-127 lin

    const int x  = blockIdx.x;   // n-tile: 64 cols per half, 45 tiles
    const int mt = blockIdx.y;   // m-tile: 128 rows, 8 tiles
    const int e  = blockIdx.z;

    const float* hs_e = hs  + (size_t)e * TOK * HID + (size_t)mt * 128 * HID;
    const float* gw_e = gup + (size_t)e * HID * TWOI;
    const float* gb_e = gub + (size_t)e * TWOI;
    unsigned short* sw_e = swbuf + (size_t)e * TOK * INTER + (size_t)mt * 128 * INTER;

    const int t   = threadIdx.x;
    const int lid = t & 31;
    const int w   = t >> 5;
    const int wm  = w & 3;            // rows wm*32
    const int wn  = w >> 2;           // cols wn*32 (within 64-col half)
    const int ncol0 = x * 64;

    v8f accg[2][2] = {};
    v8f accl[2][2] = {};

    // A staging: thread covers 4 rows, 4 consecutive K
    const int ak4 = (t & 7) * 4;
    const int am0 = t >> 3;           // 0..31
    // B staging: thread covers 4 K-rows x 4 consecutive local cols
    const int bn4 = (t & 31) * 4;     // 0..124 (local col over 128 = glu|lin)
    const int bk0 = (t >> 5) * 4;     // 0..28
    const int bcol = (bn4 < 64) ? (ncol0 + bn4) : (INTER + ncol0 + (bn4 - 64));
    const float* bsrc = gw_e + bcol;

    f32x4 pa[4], pb[4];

    auto loadA = [&](int kt) {
        #pragma unroll
        for (int i = 0; i < 4; ++i)
            pa[i] = *(const f32x4*)(hs_e + (size_t)(am0 + 32 * i) * HID + kt * BK + ak4);
    };
    auto loadB = [&](int kt) {
        #pragma unroll
        for (int i = 0; i < 4; ++i)
            pb[i] = *(const f32x4*)(bsrc + (size_t)(kt * BK + bk0 + i) * TWOI);
    };
    auto stageA = [&](unsigned short* buf) {
        #pragma unroll
        for (int i = 0; i < 4; ++i) {
            u32x2 pk;
            pk.x = cvt_pk_bf16(pa[i][0], pa[i][1]);
            pk.y = cvt_pk_bf16(pa[i][2], pa[i][3]);
            *(u32x2*)(buf + (am0 + 32 * i) * BK + ak4) = pk;
        }
    };
    auto stageB = [&](unsigned short* buf) {
        #pragma unroll
        for (int j = 0; j < 4; ++j) {
            u32x2 pk;
            pk.x = cvt_pk_bf16(pb[0][j], pb[1][j]);
            pk.y = cvt_pk_bf16(pb[2][j], pb[3][j]);
            *(u32x2*)(buf + (bn4 + j) * BK + bk0) = pk;
        }
    };
    auto compute = [&](const unsigned short* bufA, const unsigned short* bufB) {
        v16bf a0  = load_afrag(bufA, wm * 32,      lid);
        v16bf a1  = load_afrag(bufA, wm * 32 + 16, lid);
        v16bf bg0 = load_bfrag(bufB, wn * 32,      lid);
        v16bf bg1 = load_bfrag(bufB, wn * 32 + 16, lid);
        v16bf bl0 = load_bfrag(bufB, 64 + wn * 32,      lid);
        v16bf bl1 = load_bfrag(bufB, 64 + wn * 32 + 16, lid);
        accg[0][0] = WMMA_BF16(a0, bg0, accg[0][0]);
        accg[0][1] = WMMA_BF16(a0, bg1, accg[0][1]);
        accg[1][0] = WMMA_BF16(a1, bg0, accg[1][0]);
        accg[1][1] = WMMA_BF16(a1, bg1, accg[1][1]);
        accl[0][0] = WMMA_BF16(a0, bl0, accl[0][0]);
        accl[0][1] = WMMA_BF16(a0, bl1, accl[0][1]);
        accl[1][0] = WMMA_BF16(a1, bl0, accl[1][0]);
        accl[1][1] = WMMA_BF16(a1, bl1, accl[1][1]);
    };

    loadA(0); loadB(0);
    stageA(ldsA[0]); stageB(ldsB[0]);
    __syncthreads();

    int p = 0;
    for (int kt = 1; kt <= NK; ++kt) {
        const bool more = (kt < NK);
        if (more) { loadA(kt); loadB(kt); }     // prefetch next tile (global)
        compute(ldsA[p], ldsB[p]);              // WMMA on current tile
        if (more) { stageA(ldsA[p ^ 1]); stageB(ldsB[p ^ 1]); }
        __syncthreads();
        p ^= 1;
    }

    // epilogue: bias + swiglu, write bf16 intermediate
    #pragma unroll
    for (int i = 0; i < 2; ++i) {
        #pragma unroll
        for (int j = 0; j < 2; ++j) {
            int col = ncol0 + wn * 32 + j * 16 + (lid & 15);
            float bg = gb_e[col];
            float bl = gb_e[INTER + col];
            int rowbase = wm * 32 + i * 16 + ((lid >> 4) << 3);
            #pragma unroll
            for (int r = 0; r < 8; ++r) {
                float g = accg[i][j][r] + bg;
                float l = accl[i][j][r] + bl;
                float s = (g / (1.0f + __expf(-ALPHA * g))) * (l + 1.0f);
                sw_e[(size_t)(rowbase + r) * INTER + col] = f2bf(s);
            }
        }
    }
}

// ---------------------------------------------------------------------------
// Kernel 2: down GEMM + bias. Reads bf16 intermediate, writes fp32 out.
// Block: 256 thr (8 waves). Tile: 128 rows x 64 cols. Double-buffered LDS.
// A-tile (already bf16, layout-identical) is moved by the Tensor Data Mover:
// one tensor_load_to_lds per K-step issued by wave 0, drained with
// s_wait_tensorcnt before the barrier. B-tile converts fp32->bf16 in VALU.
// ---------------------------------------------------------------------------
__global__ __launch_bounds__(256)
void down_proj_kernel(const unsigned short* __restrict__ swbuf,
                      const float* __restrict__ dwn,
                      const float* __restrict__ dbi,
                      float* __restrict__ out)
{
    __shared__ __align__(16) unsigned short ldsA[2][128 * BK];
    __shared__ __align__(16) unsigned short ldsB[2][64 * BK];

    const int x  = blockIdx.x;
    const int mt = blockIdx.y;
    const int e  = blockIdx.z;

    const unsigned short* sw_e = swbuf + (size_t)e * TOK * INTER + (size_t)mt * 128 * INTER;
    const float* dw_e = dwn + (size_t)e * INTER * HID;
    const float* db_e = dbi + (size_t)e * INTER;     // broadcast over out cols (HID==INTER)
    float* out_e = out + (size_t)e * TOK * HID + (size_t)mt * 128 * HID;

    const int t   = threadIdx.x;
    const int lid = t & 31;
    const int w   = t >> 5;
    const int wm  = w & 3;
    const int wn  = w >> 2;
    const int ncol0 = x * 64;

    v8f acc[2][2] = {};

    // B staging: thread covers 2 K-rows x 4 consecutive cols
    const int bn4 = (t & 15) * 4;     // 0..60
    const int bk0 = (t >> 4) * 2;     // 0..30
    const float* bsrc = dw_e + ncol0 + bn4;

    f32x4 pb[2];

    auto tdmA = [&](int kt, unsigned short* buf) {
        if (w == 0) {
            tdm_load_tile_bf16(sw_e + kt * BK,
                               (unsigned)(uintptr_t)(void*)buf,
                               INTER, 128u, (unsigned)BK);
        }
    };
    auto loadB = [&](int kt) {
        #pragma unroll
        for (int i = 0; i < 2; ++i)
            pb[i] = *(const f32x4*)(bsrc + (size_t)(kt * BK + bk0 + i) * HID);
    };
    auto stageB = [&](unsigned short* buf) {
        #pragma unroll
        for (int j = 0; j < 4; ++j) {
            unsigned int pk = cvt_pk_bf16(pb[0][j], pb[1][j]);
            *(unsigned int*)(buf + (bn4 + j) * BK + bk0) = pk;
        }
    };
    auto compute = [&](const unsigned short* bufA, const unsigned short* bufB) {
        v16bf a0 = load_afrag(bufA, wm * 32,      lid);
        v16bf a1 = load_afrag(bufA, wm * 32 + 16, lid);
        v16bf b0 = load_bfrag(bufB, wn * 32,      lid);
        v16bf b1 = load_bfrag(bufB, wn * 32 + 16, lid);
        acc[0][0] = WMMA_BF16(a0, b0, acc[0][0]);
        acc[0][1] = WMMA_BF16(a0, b1, acc[0][1]);
        acc[1][0] = WMMA_BF16(a1, b0, acc[1][0]);
        acc[1][1] = WMMA_BF16(a1, b1, acc[1][1]);
    };

    // prologue: stage tile 0
    tdmA(0, ldsA[0]);
    loadB(0);
    stageB(ldsB[0]);
    if (w == 0) __builtin_amdgcn_s_wait_tensorcnt(0);
    __syncthreads();

    int p = 0;
    for (int kt = 1; kt <= NK; ++kt) {
        const bool more = (kt < NK);
        if (more) { tdmA(kt, ldsA[p ^ 1]); loadB(kt); }  // TDM + global prefetch
        compute(ldsA[p], ldsB[p]);                       // WMMA on current tile
        if (more) { stageB(ldsB[p ^ 1]); }
        if (w == 0) __builtin_amdgcn_s_wait_tensorcnt(0);
        __syncthreads();
        p ^= 1;
    }

    // epilogue: bias, fp32 store
    #pragma unroll
    for (int i = 0; i < 2; ++i) {
        #pragma unroll
        for (int j = 0; j < 2; ++j) {
            int col = ncol0 + wn * 32 + j * 16 + (lid & 15);
            float b = db_e[col];
            int rowbase = wm * 32 + i * 16 + ((lid >> 4) << 3);
            #pragma unroll
            for (int r = 0; r < 8; ++r) {
                out_e[(size_t)(rowbase + r) * HID + col] = acc[i][j][r] + b;
            }
        }
    }
}

extern "C" void kernel_launch(void* const* d_in, const int* in_sizes, int n_in,
                              void* d_out, int out_size, void* d_ws, size_t ws_size,
                              hipStream_t stream) {
    const float* hs  = (const float*)d_in[0];   // [E*T, H]
    const float* gup = (const float*)d_in[1];   // [E, H, 2I]
    const float* gub = (const float*)d_in[2];   // [E, 2I]
    const float* dwn = (const float*)d_in[3];   // [E, I, H]
    const float* dbi = (const float*)d_in[4];   // [E, I]
    float* out = (float*)d_out;                 // [E*T, H] fp32
    unsigned short* swbuf = (unsigned short*)d_ws;  // bf16 intermediate [E, T, I] = 47 MB

    dim3 block(256);
    dim3 gridA(INTER / 64, TOK / 128, EXPERTS);   // 45 x 8 x 8
    gate_up_swiglu_kernel<<<gridA, block, 0, stream>>>(hs, gup, gub, swbuf);

    dim3 gridB(HID / 64, TOK / 128, EXPERTS);     // 45 x 8 x 8
    down_proj_kernel<<<gridB, block, 0, stream>>>(swbuf, dwn, dbi, out);
}
